// PointTransformer_fp_59897613910172
// MI455X (gfx1250) — compile-verified, matrix operand order checked by reading, not run on hardware
//
#include <hip/hip_runtime.h>
#include <hip/hip_bf16.h>

typedef __attribute__((ext_vector_type(16))) _Float16 v16h;
typedef __attribute__((ext_vector_type(8)))  float    v8f;

#define CDIV(a,b) (((a)+(b)-1)/(b))
#define WMMA(a,b,c) __builtin_amdgcn_wmma_f32_16x16x32_f16(false,(a),false,(b),(short)0,(c),false,false)

// ---------------------------------------------------------------------------
// f32 -> f16 conversion
// ---------------------------------------------------------------------------
__global__ void k_cvt(const float* __restrict__ s, _Float16* __restrict__ d, long n) {
    long i = (long)blockIdx.x * blockDim.x + threadIdx.x;
    if (i < n) d[i] = (_Float16)s[i];
}

// ---------------------------------------------------------------------------
// First layer: out = relu(bn( x[.,3] @ w1[64,3]^T ))   (K=3 too small for WMMA)
// ---------------------------------------------------------------------------
__global__ void k_dense3(const float* __restrict__ x, const float* __restrict__ w,
                         const float* __restrict__ g, const float* __restrict__ b,
                         _Float16* __restrict__ out, long total) {
    long i = (long)blockIdx.x * blockDim.x + threadIdx.x;
    if (i >= total) return;
    long m = i >> 6; int c = (int)(i & 63);
    const float* xp = x + m * 3;
    float acc = xp[0]*w[c*3] + xp[1]*w[c*3+1] + xp[2]*w[c*3+2];
    acc = acc * g[c] + b[c];
    out[i] = (_Float16)fmaxf(acc, 0.f);
}

// ---------------------------------------------------------------------------
// WMMA GEMM:  out[b,m,n] = act( (sum_k A[b,m,k]*W[b,n,k] + bias[n]) * g[n] + be[n] )
// A: f16 [.,K] row-major, W: f16 [Nd,K] row-major (i.e. computes A @ W^T).
// One wave per 32x32 C tile (2x2 of 16x16), 4 waves/block along N.
// Manual software pipeline: next K-step fragments loaded before current WMMAs.
// act: 0=none 1=relu 2=leaky(0.2)
// ---------------------------------------------------------------------------
__device__ __forceinline__ void k_gemm_store(float acc, int row, int col, int M, int Nd,
                                             float bi, float g, float be, int act,
                                             _Float16* out16, float* out32, long base) {
    if (row < M && col < Nd) {
        float y = acc + bi;
        y = y * g + be;
        if (act == 1) y = fmaxf(y, 0.f);
        else if (act == 2) y = y > 0.f ? y : 0.2f * y;
        long o = base + (long)row * Nd + col;
        if (out16) out16[o] = (_Float16)y;
        if (out32) out32[o] = y;
    }
}

__global__ void k_gemm(const _Float16* __restrict__ A, const _Float16* __restrict__ W,
                       _Float16* __restrict__ out16, float* __restrict__ out32,
                       const float* __restrict__ bias, const float* __restrict__ gamma,
                       const float* __restrict__ beta,
                       int M, int Nd, int K, long sA, long sW, long sO, int act) {
    int wave = threadIdx.x >> 5;
    int lane = threadIdx.x & 31;
    int tileM = blockIdx.x * 32;
    int tileN = (blockIdx.y * 4 + wave) * 32;
    if (tileN >= Nd) return;                          // wave-uniform
    int bz = blockIdx.z;
    const _Float16* Ab = A + (long)bz * sA;
    const _Float16* Wb = W + (long)bz * sW;

    int half = lane >> 4;
    int mr   = lane & 15;
    int rA0 = tileM + mr;      if (rA0 >= M)  rA0 = M - 1;
    int rA1 = tileM + 16 + mr; if (rA1 >= M)  rA1 = M - 1;
    int rW0 = tileN + mr;      if (rW0 >= Nd) rW0 = Nd - 1;
    int rW1 = tileN + 16 + mr; if (rW1 >= Nd) rW1 = Nd - 1;
    const _Float16* pa0 = Ab + (long)rA0 * K + half * 16;
    const _Float16* pa1 = Ab + (long)rA1 * K + half * 16;
    const _Float16* pw0 = Wb + (long)rW0 * K + half * 16;
    const _Float16* pw1 = Wb + (long)rW1 * K + half * 16;

    v8f c00 = {}, c01 = {}, c10 = {}, c11 = {};
    v16h a0 = *(const v16h*)pa0;
    v16h a1 = *(const v16h*)pa1;
    v16h b0 = *(const v16h*)pw0;
    v16h b1 = *(const v16h*)pw1;

    for (int k = 32; k < K; k += 32) {
        v16h na0 = *(const v16h*)(pa0 + k);
        v16h na1 = *(const v16h*)(pa1 + k);
        v16h nb0 = *(const v16h*)(pw0 + k);
        v16h nb1 = *(const v16h*)(pw1 + k);
        if (k + 64 < K) {
            __builtin_prefetch(pa0 + k + 64, 0, 3);
            __builtin_prefetch(pa1 + k + 64, 0, 3);
        }
        c00 = WMMA(a0, b0, c00);
        c01 = WMMA(a0, b1, c01);
        c10 = WMMA(a1, b0, c10);
        c11 = WMMA(a1, b1, c11);
        a0 = na0; a1 = na1; b0 = nb0; b1 = nb1;
    }
    c00 = WMMA(a0, b0, c00);
    c01 = WMMA(a0, b1, c01);
    c10 = WMMA(a1, b0, c10);
    c11 = WMMA(a1, b1, c11);

    int col0 = tileN + mr, col1 = tileN + 16 + mr;
    float g0 = 1.f, be0 = 0.f, bi0 = 0.f, g1 = 1.f, be1 = 0.f, bi1 = 0.f;
    if (col0 < Nd) {
        if (gamma) { g0 = gamma[col0]; be0 = beta[col0]; }
        if (bias)  bi0 = bias[col0];
    }
    if (col1 < Nd) {
        if (gamma) { g1 = gamma[col1]; be1 = beta[col1]; }
        if (bias)  bi1 = bias[col1];
    }
    long base = (long)bz * sO;
#pragma unroll
    for (int r = 0; r < 8; ++r) {
        int row0 = tileM + r + half * 8;
        int row1 = row0 + 16;
        k_gemm_store(c00[r], row0, col0, M, Nd, bi0, g0, be0, act, out16, out32, base);
        k_gemm_store(c01[r], row0, col1, M, Nd, bi1, g1, be1, act, out16, out32, base);
        k_gemm_store(c10[r], row1, col0, M, Nd, bi0, g0, be0, act, out16, out32, base);
        k_gemm_store(c11[r], row1, col1, M, Nd, bi1, g1, be1, act, out16, out32, base);
    }
}

// ---------------------------------------------------------------------------
// Farthest point sampling (one block per batch, dist array in LDS)
// ---------------------------------------------------------------------------
__global__ void k_fps(const float* __restrict__ xyz, int* __restrict__ idx,
                      int n, int npoint) {
    __shared__ float dist[4096];
    __shared__ float sv[256]; __shared__ int si[256];
    __shared__ int curS;
    int b = blockIdx.x, t = threadIdx.x;
    const float* p = xyz + (long)b * n * 3;
    for (int i = t; i < n; i += 256) dist[i] = 1e10f;
    if (t == 0) { idx[(long)b * npoint] = 0; curS = 0; }
    __syncthreads();
    for (int s = 1; s < npoint; ++s) {
        int cur = curS;
        float cx = p[cur*3], cy = p[cur*3+1], cz = p[cur*3+2];
        float bv = -1.f; int bi = 0x7fffffff;
        for (int i = t; i < n; i += 256) {
            float dx = p[i*3]-cx, dy = p[i*3+1]-cy, dz = p[i*3+2]-cz;
            float d = dx*dx + dy*dy + dz*dz;
            float dm = fminf(dist[i], d);
            dist[i] = dm;
            if (dm > bv || (dm == bv && i < bi)) { bv = dm; bi = i; }
        }
        sv[t] = bv; si[t] = bi;
        __syncthreads();
        for (int o = 128; o > 0; o >>= 1) {
            if (t < o) {
                if (sv[t+o] > sv[t] || (sv[t+o] == sv[t] && si[t+o] < si[t])) {
                    sv[t] = sv[t+o]; si[t] = si[t+o];
                }
            }
            __syncthreads();
        }
        if (t == 0) { curS = si[0]; idx[(long)b * npoint + s] = si[0]; }
        __syncthreads();
    }
}

__global__ void k_gather3(const float* __restrict__ xyz, const int* __restrict__ idx,
                          float* __restrict__ out, int n, int s_, long total) {
    long i = (long)blockIdx.x * blockDim.x + threadIdx.x;
    if (i >= total) return;
    int c = (int)(i % 3); long r = i / 3;
    int s = (int)(r % s_); int b = (int)(r / s_);
    out[i] = xyz[((long)b * n + idx[(long)b * s_ + s]) * 3 + c];
}

// ---------------------------------------------------------------------------
// 32-NN: one block (128 thr) per query; LDS distance array + iterated argmin
// ---------------------------------------------------------------------------
__global__ void k_knn(const float* __restrict__ qxyz, const float* __restrict__ xyz,
                      int* __restrict__ knn, int n, int s_, int kk) {
    __shared__ float d[4096];
    __shared__ unsigned sel[128];
    __shared__ float sv[128]; __shared__ int si[128];
    int q = blockIdx.x;                 // b*s_ + s
    int b = q / s_;
    int t = threadIdx.x;
    float qx = qxyz[q*3], qy = qxyz[q*3+1], qz = qxyz[q*3+2];
    const float* p = xyz + (long)b * n * 3;
    for (int i = t; i < n; i += 128) {
        float dx = p[i*3]-qx, dy = p[i*3+1]-qy, dz = p[i*3+2]-qz;
        d[i] = dx*dx + dy*dy + dz*dz;
    }
    sel[t] = 0;
    __syncthreads();
    for (int k = 0; k < kk; ++k) {
        float bv = 3.4e38f; int bi = 0x7fffffff;
        for (int i = t; i < n; i += 128) {
            if (!((sel[i >> 5] >> (i & 31)) & 1u)) {
                float dv = d[i];
                if (dv < bv || (dv == bv && i < bi)) { bv = dv; bi = i; }
            }
        }
        sv[t] = bv; si[t] = bi;
        __syncthreads();
        for (int o = 64; o > 0; o >>= 1) {
            if (t < o) {
                if (sv[t+o] < sv[t] || (sv[t+o] == sv[t] && si[t+o] < si[t])) {
                    sv[t] = sv[t+o]; si[t] = si[t+o];
                }
            }
            __syncthreads();
        }
        if (t == 0) { knn[(long)q * kk + k] = si[0]; sel[si[0] >> 5] |= 1u << (si[0] & 31); }
        __syncthreads();
    }
}

// ---------------------------------------------------------------------------
// Build grouped features: out[b,s,k, 0:C]=pts[knn]-center, out[b,s,k, C:2C]=center
// ---------------------------------------------------------------------------
__global__ void k_group(const _Float16* __restrict__ pts, const int* __restrict__ fpsIdx,
                        const int* __restrict__ knnIdx, _Float16* __restrict__ out,
                        int n, int s_, int kk, int C, long total) {
    long i = (long)blockIdx.x * blockDim.x + threadIdx.x;
    if (i >= total) return;
    int c2 = (int)(i % (2*C)); long r = i / (2*C);
    int k = (int)(r % kk); r /= kk;
    int s = (int)(r % s_); int b = (int)(r / s_);
    int center = fpsIdx[(long)b * s_ + s];
    float cen = (float)pts[((long)b * n + center) * C + (c2 % C)];
    if (c2 < C) {
        int g = knnIdx[((long)b * s_ + s) * kk + k];
        out[i] = (_Float16)((float)pts[((long)b * n + g) * C + c2] - cen);
    } else {
        out[i] = (_Float16)cen;
    }
}

__global__ void k_maxk(const _Float16* __restrict__ in, _Float16* __restrict__ out,
                       int kk, int C, long total) {
    long i = (long)blockIdx.x * blockDim.x + threadIdx.x;
    if (i >= total) return;
    long s = i / C; int c = (int)(i % C);
    const _Float16* p = in + s * (long)kk * C + c;
    float m = (float)p[0];
    for (int k = 1; k < kk; ++k) m = fmaxf(m, (float)p[(long)k * C]);
    out[i] = (_Float16)m;
}

// ---------------------------------------------------------------------------
// Attention helpers
// ---------------------------------------------------------------------------
__global__ void k_softmax(float* __restrict__ e, int cols) {
    __shared__ float red[256];
    long row = blockIdx.x;
    float* p = e + row * (long)cols;
    int t = threadIdx.x;
    float mx = -3.4e38f;
    for (int i = t; i < cols; i += 256) mx = fmaxf(mx, p[i]);
    red[t] = mx; __syncthreads();
    for (int o = 128; o > 0; o >>= 1) { if (t < o) red[t] = fmaxf(red[t], red[t+o]); __syncthreads(); }
    mx = red[0]; __syncthreads();
    float sm = 0.f;
    for (int i = t; i < cols; i += 256) { float v = __expf(p[i] - mx); p[i] = v; sm += v; }
    red[t] = sm; __syncthreads();
    for (int o = 128; o > 0; o >>= 1) { if (t < o) red[t] += red[t+o]; __syncthreads(); }
    float inv = 1.f / red[0];
    for (int i = t; i < cols; i += 256) p[i] *= inv;
}

__global__ void k_colsum(const float* __restrict__ attn, float* __restrict__ cs,
                         int M_, long total) {
    long i = (long)blockIdx.x * blockDim.x + threadIdx.x;
    if (i >= total) return;
    int b = (int)(i / M_); int j = (int)(i % M_);
    const float* p = attn + (long)b * M_ * M_ + j;
    float s = 0.f;
    for (int r = 0; r < M_; ++r) s += p[(long)r * M_];
    cs[i] = s;
}

// attnT[b,j,i] = attn[b,i,j] / (1e-9 + colsum[b,j])   (f16 out)
__global__ void k_attnT(const float* __restrict__ attn, const float* __restrict__ cs,
                        _Float16* __restrict__ at, int M_, long total) {
    long i = (long)blockIdx.x * blockDim.x + threadIdx.x;
    if (i >= total) return;
    int ii = (int)(i % M_); long r = i / M_;
    int j = (int)(r % M_); int b = (int)(r / M_);
    float v = attn[((long)b * M_ + ii) * M_ + j];
    at[i] = (_Float16)(v / (1e-9f + cs[(long)b * M_ + j]));
}

__global__ void k_tr(const _Float16* __restrict__ src, _Float16* __restrict__ dst,
                     int R, int C, long total) {
    long i = (long)blockIdx.x * blockDim.x + threadIdx.x;
    if (i >= total) return;
    int c = (int)(i % C); long r = i / C;
    int rr = (int)(r % R); int b = (int)(r / R);
    dst[((long)b * C + c) * R + rr] = src[i];
}

__global__ void k_sub(const _Float16* __restrict__ a, const _Float16* __restrict__ b,
                      _Float16* __restrict__ o, long total) {
    long i = (long)blockIdx.x * blockDim.x + threadIdx.x;
    if (i >= total) return;
    o[i] = (_Float16)((float)a[i] - (float)b[i]);
}

// out = a + b (contiguous); also scatter into cat[r*ldo + off + c] when cat != 0
__global__ void k_add2(const _Float16* __restrict__ a, const _Float16* __restrict__ b,
                       _Float16* __restrict__ out, _Float16* __restrict__ cat,
                       int C, int ldo, int off, long total) {
    long i = (long)blockIdx.x * blockDim.x + threadIdx.x;
    if (i >= total) return;
    long r = i / C; int c = (int)(i % C);
    float v = (float)a[i] + (float)b[i];
    out[i] = (_Float16)v;
    if (cat) cat[r * (long)ldo + off + c] = (_Float16)v;
}

__global__ void k_cpys(const _Float16* __restrict__ src, _Float16* __restrict__ dst,
                       int C, int ldo, int off, long total) {
    long i = (long)blockIdx.x * blockDim.x + threadIdx.x;
    if (i >= total) return;
    long r = i / C; int c = (int)(i % C);
    dst[r * (long)ldo + off + c] = src[i];
}

__global__ void k_rowmax(const _Float16* __restrict__ h, _Float16* __restrict__ hmax,
                         int Mpts, int C, long total) {
    long i = (long)blockIdx.x * blockDim.x + threadIdx.x;
    if (i >= total) return;
    int b = (int)(i / C); int c = (int)(i % C);
    const _Float16* p = h + (long)b * Mpts * C + c;
    float m = (float)p[0];
    for (int r = 1; r < Mpts; ++r) m = fmaxf(m, (float)p[(long)r * C]);
    hmax[i] = (_Float16)m;
}

__global__ void k_bcast(const _Float16* __restrict__ hmax, _Float16* __restrict__ dst,
                        int C, int ldo, int off, int Mpts, long total) {
    long i = (long)blockIdx.x * blockDim.x + threadIdx.x;
    if (i >= total) return;
    long r = i / C; int c = (int)(i % C);
    int b = (int)(r / Mpts);
    dst[r * (long)ldo + off + c] = hmax[(long)b * C + c];
}

// ---------------------------------------------------------------------------
// 3-NN inverse-distance interpolation writing into a concat buffer
// ---------------------------------------------------------------------------
__global__ void k_fpinterp(const float* __restrict__ dxyz, const float* __restrict__ sxyz,
                           const _Float16* __restrict__ pts2, _Float16* __restrict__ dst,
                           int ndense, int nsparse, int C2, int ldo, int off) {
    __shared__ float d[2048];
    __shared__ float sv[128]; __shared__ int si[128];
    __shared__ int nn[3]; __shared__ float wgt[3];
    int q = blockIdx.x;              // b*ndense + s
    int b = q / ndense;
    int t = threadIdx.x;
    float qx = dxyz[q*3], qy = dxyz[q*3+1], qz = dxyz[q*3+2];
    const float* sp = sxyz + (long)b * nsparse * 3;
    for (int i = t; i < nsparse; i += 128) {
        float dx = sp[i*3]-qx, dy = sp[i*3+1]-qy, dz = sp[i*3+2]-qz;
        d[i] = dx*dx + dy*dy + dz*dz;
    }
    __syncthreads();
    for (int k = 0; k < 3; ++k) {
        float bv = 3.4e38f; int bi = 0x7fffffff;
        for (int i = t; i < nsparse; i += 128) {
            float dv = d[i];
            if (dv < bv || (dv == bv && i < bi)) { bv = dv; bi = i; }
        }
        sv[t] = bv; si[t] = bi;
        __syncthreads();
        for (int o = 64; o > 0; o >>= 1) {
            if (t < o) {
                if (sv[t+o] < sv[t] || (sv[t+o] == sv[t] && si[t+o] < si[t])) {
                    sv[t] = sv[t+o]; si[t] = si[t+o];
                }
            }
            __syncthreads();
        }
        if (t == 0) { nn[k] = si[0]; wgt[k] = 1.f / (sv[0] + 1e-8f); d[si[0]] = 3.4e38f; }
        __syncthreads();
    }
    if (t == 0) { float s = wgt[0]+wgt[1]+wgt[2]; wgt[0] /= s; wgt[1] /= s; wgt[2] /= s; }
    __syncthreads();
    const _Float16* p0 = pts2 + ((long)b * nsparse + nn[0]) * C2;
    const _Float16* p1 = pts2 + ((long)b * nsparse + nn[1]) * C2;
    const _Float16* p2 = pts2 + ((long)b * nsparse + nn[2]) * C2;
    _Float16* dp = dst + (long)q * ldo + off;
    float w0 = wgt[0], w1 = wgt[1], w2 = wgt[2];
    for (int c = t; c < C2; c += 128)
        dp[c] = (_Float16)(w0*(float)p0[c] + w1*(float)p1[c] + w2*(float)p2[c]);
}

// ---------------------------------------------------------------------------
// Host side
// ---------------------------------------------------------------------------
static void gemm(hipStream_t st, const _Float16* A, const _Float16* W,
                 _Float16* o16, float* o32, const float* bias,
                 const float* gamma, const float* beta,
                 int M, int Nd, int K, long sA, long sW, long sO, int act, int batch) {
    dim3 grid(CDIV(M, 32), CDIV(Nd, 128), batch);
    k_gemm<<<grid, 128, 0, st>>>(A, W, o16, o32, bias, gamma, beta,
                                 M, Nd, K, sA, sW, sO, act);
}

static unsigned gsz(long total) { return (unsigned)CDIV(total, 256); }

// Input leaf order (jax pytree: sorted dict keys, list order preserved):
//  0 b6  1 b7  2 bl  3 bn1.b  4 bn1.g  5 bn2.b  6 bn2.g  7 bn6.b  8 bn6.g
//  9 bn7.b 10 bn7.g
// 11 fp1.b0 12 fp1.b1 13 fp1.bn0.b 14 fp1.bn0.g 15 fp1.bn1.b 16 fp1.bn1.g
// 17 fp1.w0[1024,2176] 18 fp1.w1[512,1024]
// 19 fp2.b0 20 fp2.b1 21 fp2.bn0.b 22 fp2.bn0.g 23 fp2.bn1.b 24 fp2.bn1.g
// 25 fp2.w0[256,576] 26 fp2.w1[128,256]
// 27 fuse.bn.b 28 fuse.bn.g 29 fuse.w[2048,2048]
// 30 gl1.bn1.b 31 gl1.bn1.g 32 gl1.bn2.b 33 gl1.bn2.g 34 gl1.w1 35 gl1.w2
// 36 gl2.bn1.b 37 gl2.bn1.g 38 gl2.bn2.b 39 gl2.bn2.g 40 gl2.w1 41 gl2.w2
// 42 pt.bn1.b 43 pt.bn1.g 44 pt.bn2.b 45 pt.bn2.g
// 46+7l: sa[l].{b, bt, bv, g, wq[64,256], wt[256,256], wv[256,256]}
// 74 pt.w1 75 pt.w2 76 w1[64,3] 77 w2[64,64] 78 w6[64,128] 79 w7[32,64]
// 80 wl[8,32] 81 x[8,4096,3]
extern "C" void kernel_launch(void* const* d_in, const int* in_sizes, int n_in,
                              void* d_out, int out_size, void* d_ws, size_t ws_size,
                              hipStream_t stream) {
    (void)in_sizes; (void)n_in; (void)out_size; (void)ws_size;
    const int Bn = 8, Npts = 4096, S1 = 2048, S2 = 1024, KNN = 32;
    const float* x = (const float*)d_in[81];

    char* wsp = (char*)d_ws;
    auto alloc = [&](size_t bytes) -> void* {
        void* r = wsp; wsp += (bytes + 255) & ~(size_t)255; return r;
    };
    auto cvtw = [&](int idx, long n) -> _Float16* {
        _Float16* dd = (_Float16*)alloc((size_t)n * sizeof(_Float16));
        k_cvt<<<gsz(n), 256, 0, stream>>>((const float*)d_in[idx], dd, n);
        return dd;
    };
    auto F = [&](int idx) -> const float* { return (const float*)d_in[idx]; };

    // ---- weight conversions (f16) ----
    _Float16* w2f   = cvtw(77, 64*64);
    _Float16* g1w1  = cvtw(34, 128*128);
    _Float16* g1w2  = cvtw(35, 128*128);
    _Float16* g2w1  = cvtw(40, 256*256);
    _Float16* g2w2  = cvtw(41, 256*256);
    _Float16* ptw1  = cvtw(74, 256*256);
    _Float16* ptw2  = cvtw(75, 256*256);
    _Float16* saq[4]; _Float16* sat[4]; _Float16* sav[4];
    for (int l = 0; l < 4; ++l) {
        int sb = 46 + 7*l;
        saq[l] = cvtw(sb+4, 64*256);
        sat[l] = cvtw(sb+5, 256*256);
        sav[l] = cvtw(sb+6, 256*256);
    }
    _Float16* fusew = cvtw(29, 2048*2048);
    _Float16* fp1w0 = cvtw(17, 1024*2176);
    _Float16* fp1w1 = cvtw(18, 512*1024);
    _Float16* fp2w0 = cvtw(25, 256*576);
    _Float16* fp2w1 = cvtw(26, 128*256);
    _Float16* w6f   = cvtw(78, 64*128);
    _Float16* w7f   = cvtw(79, 32*64);
    _Float16* wlf   = cvtw(80, 8*32);

    // ---- activation buffers ----
    long nBN = (long)Bn * Npts;                 // 32768 rows
    _Float16* f0a  = (_Float16*)alloc(nBN*64*2);
    _Float16* f0   = (_Float16*)alloc(nBN*64*2);
    int*   fpsIdx1 = (int*)  alloc((long)Bn*S1*4);
    float* xyz2    = (float*)alloc((long)Bn*S1*3*4);
    int*   knn1    = (int*)  alloc((long)Bn*S1*KNN*4);
    int*   fpsIdx2 = (int*)  alloc((long)Bn*S2*4);
    float* xyz3    = (float*)alloc((long)Bn*S2*3*4);
    int*   knn2    = (int*)  alloc((long)Bn*S2*KNN*4);
    _Float16* f1   = (_Float16*)alloc((long)Bn*S1*128*2);
    _Float16* f2   = (_Float16*)alloc((long)Bn*S2*256*2);
    _Float16* GA   = (_Float16*)alloc(67108864L*2);   // big arena A (134 MB)
    _Float16* GB   = (_Float16*)alloc(67108864L*2);   // big arena B (134 MB)
    _Float16* pth1 = (_Float16*)alloc((long)Bn*S2*256*2);
    _Float16* hA   = (_Float16*)alloc((long)Bn*S2*256*2);
    _Float16* hB   = (_Float16*)alloc((long)Bn*S2*256*2);
    _Float16* qbuf = (_Float16*)alloc((long)Bn*S2*64*2);
    float* energy  = (float*)alloc((long)Bn*S2*S2*4);
    float* colsum  = (float*)alloc((long)Bn*S2*4);
    _Float16* attnT= (_Float16*)alloc((long)Bn*S2*S2*2);
    _Float16* vbuf = (_Float16*)alloc((long)Bn*S2*256*2);
    _Float16* vT   = (_Float16*)alloc((long)Bn*S2*256*2);
    _Float16* x_r  = (_Float16*)alloc((long)Bn*S2*256*2);
    _Float16* tsub = (_Float16*)alloc((long)Bn*S2*256*2);
    _Float16* xr2  = (_Float16*)alloc((long)Bn*S2*256*2);
    _Float16* sacat= (_Float16*)alloc((long)Bn*S2*1024*2);
    _Float16* hmaxb= (_Float16*)alloc((long)Bn*1024*2);
    _Float16* fuseo= (_Float16*)alloc((long)Bn*S2*2048*2);
    _Float16* fp1a = (_Float16*)alloc((long)Bn*S1*1024*2);
    _Float16* fp1b = (_Float16*)alloc((long)Bn*S1*512*2);
    _Float16* fp2a = (_Float16*)alloc(nBN*256*2);
    _Float16* fp2b = (_Float16*)alloc(nBN*128*2);
    _Float16* h6   = (_Float16*)alloc(nBN*64*2);
    _Float16* h6f  = (_Float16*)alloc(nBN*64*2);
    _Float16* h7   = (_Float16*)alloc(nBN*32*2);

    // ================= pipeline =================
    // f0a = relu(bn1(x @ w1^T)); f0 = relu(bn2(f0a @ w2^T))
    k_dense3<<<gsz(nBN*64), 256, 0, stream>>>(x, F(76), F(4), F(3), f0a, nBN*64);
    gemm(stream, f0a, w2f, f0, nullptr, nullptr, F(6), F(5),
         (int)nBN, 64, 64, 0, 0, 0, 1, 1);

    // ---- level 1 sample+group ----
    k_fps<<<Bn, 256, 0, stream>>>(x, fpsIdx1, Npts, S1);
    k_gather3<<<gsz((long)Bn*S1*3), 256, 0, stream>>>(x, fpsIdx1, xyz2, Npts, S1, (long)Bn*S1*3);
    k_knn<<<Bn*S1, 128, 0, stream>>>(xyz2, x, knn1, Npts, S1, KNN);
    long tg1 = (long)Bn*S1*KNN*128;
    k_group<<<gsz(tg1), 256, 0, stream>>>(f0, fpsIdx1, knn1, GA, Npts, S1, KNN, 64, tg1);
    // local_op gl1
    int Mg1 = Bn*S1*KNN;   // 524288 rows
    gemm(stream, GA, g1w1, GB, nullptr, nullptr, F(31), F(30), Mg1, 128, 128, 0,0,0, 1, 1);
    gemm(stream, GB, g1w2, GA, nullptr, nullptr, F(33), F(32), Mg1, 128, 128, 0,0,0, 1, 1);
    k_maxk<<<gsz((long)Bn*S1*128), 256, 0, stream>>>(GA, f1, KNN, 128, (long)Bn*S1*128);

    // ---- level 2 sample+group ----
    k_fps<<<Bn, 256, 0, stream>>>(xyz2, fpsIdx2, S1, S2);
    k_gather3<<<gsz((long)Bn*S2*3), 256, 0, stream>>>(xyz2, fpsIdx2, xyz3, S1, S2, (long)Bn*S2*3);
    k_knn<<<Bn*S2, 128, 0, stream>>>(xyz3, xyz2, knn2, S1, S2, KNN);
    long tg2 = (long)Bn*S2*KNN*256;
    k_group<<<gsz(tg2), 256, 0, stream>>>(f1, fpsIdx2, knn2, GA, S1, S2, KNN, 128, tg2);
    int Mg2 = Bn*S2*KNN;   // 262144 rows
    gemm(stream, GA, g2w1, GB, nullptr, nullptr, F(37), F(36), Mg2, 256, 256, 0,0,0, 1, 1);
    gemm(stream, GB, g2w2, GA, nullptr, nullptr, F(39), F(38), Mg2, 256, 256, 0,0,0, 1, 1);
    k_maxk<<<gsz((long)Bn*S2*256), 256, 0, stream>>>(GA, f2, KNN, 256, (long)Bn*S2*256);

    // ---- stacked attention ----
    int Ma = Bn * S2;      // 8192 rows
    gemm(stream, f2,   ptw1, pth1, nullptr, nullptr, F(43), F(42), Ma, 256, 256, 0,0,0, 1, 1);
    gemm(stream, pth1, ptw2, hA,   nullptr, nullptr, F(45), F(44), Ma, 256, 256, 0,0,0, 1, 1);
    _Float16* hcur = hA; _Float16* hnxt = hB;
    long tE = (long)Bn*S2*S2;
    for (int l = 0; l < 4; ++l) {
        int sb = 46 + 7*l;
        gemm(stream, hcur, saq[l], qbuf, nullptr, nullptr, nullptr, nullptr,
             Ma, 64, 256, 0, 0, 0, 0, 1);
        // energy[b] = q[b] @ q[b]^T  (batched, f32 out)
        gemm(stream, qbuf, qbuf, nullptr, energy, nullptr, nullptr, nullptr,
             S2, S2, 64, (long)S2*64, (long)S2*64, (long)S2*S2, 0, Bn);
        k_softmax<<<Bn*S2, 256, 0, stream>>>(energy, S2);
        k_colsum<<<gsz((long)Bn*S2), 256, 0, stream>>>(energy, colsum, S2, (long)Bn*S2);
        k_attnT<<<gsz(tE), 256, 0, stream>>>(energy, colsum, attnT, S2, tE);
        gemm(stream, hcur, sav[l], vbuf, nullptr, F(sb+2), nullptr, nullptr,
             Ma, 256, 256, 0, 0, 0, 0, 1);
        k_tr<<<gsz((long)Ma*256), 256, 0, stream>>>(vbuf, vT, S2, 256, (long)Ma*256);
        // x_r[b] = attnT[b] @ v[b]  (W = vT[b])
        gemm(stream, attnT, vT, x_r, nullptr, nullptr, nullptr, nullptr,
             S2, 256, S2, (long)S2*S2, (long)256*S2, (long)S2*256, 0, Bn);
        k_sub<<<gsz((long)Ma*256), 256, 0, stream>>>(hcur, x_r, tsub, (long)Ma*256);
        gemm(stream, tsub, sat[l], xr2, nullptr, F(sb+1), F(sb+3), F(sb+0),
             Ma, 256, 256, 0, 0, 0, 1, 1);
        k_add2<<<gsz((long)Ma*256), 256, 0, stream>>>(hcur, xr2, hnxt, sacat,
                                                      256, 1024, l*256, (long)Ma*256);
        _Float16* tmp = hcur; hcur = hnxt; hnxt = tmp;
    }

    // ---- fuse: concat [sacat, broadcast max] -> leaky-relu dense ----
    _Float16* fusein = GB;       // arena reuse (rows 8192 x 2048)
    k_rowmax<<<gsz((long)Bn*1024), 256, 0, stream>>>(sacat, hmaxb, S2, 1024, (long)Bn*1024);
    k_cpys <<<gsz((long)Ma*1024), 256, 0, stream>>>(sacat, fusein, 1024, 2048, 0, (long)Ma*1024);
    k_bcast<<<gsz((long)Ma*1024), 256, 0, stream>>>(hmaxb, fusein, 1024, 2048, 1024, S2, (long)Ma*1024);
    gemm(stream, fusein, fusew, fuseo, nullptr, nullptr, F(28), F(27),
         Ma, 2048, 2048, 0, 0, 0, 2, 1);

    // ---- feature propagation 1: (xyz2 dense, xyz3 sparse) ----
    _Float16* fp1cat = GA;       // rows Bn*S1, ld 2176
    k_cpys<<<gsz((long)Bn*S1*128), 256, 0, stream>>>(f1, fp1cat, 128, 2176, 0, (long)Bn*S1*128);
    k_fpinterp<<<Bn*S1, 128, 0, stream>>>(xyz2, xyz3, fuseo, fp1cat, S1, S2, 2048, 2176, 128);
    gemm(stream, fp1cat, fp1w0, fp1a, nullptr, F(11), F(14), F(13),
         Bn*S1, 1024, 2176, 0, 0, 0, 1, 1);
    gemm(stream, fp1a, fp1w1, fp1b, nullptr, F(12), F(16), F(15),
         Bn*S1, 512, 1024, 0, 0, 0, 1, 1);

    // ---- feature propagation 2: (x dense, xyz2 sparse) ----
    _Float16* fp2cat = GB;       // rows Bn*Npts, ld 576
    k_cpys<<<gsz(nBN*64), 256, 0, stream>>>(f0, fp2cat, 64, 576, 0, nBN*64);
    k_fpinterp<<<Bn*Npts, 128, 0, stream>>>(x, xyz2, fp1b, fp2cat, Npts, S1, 512, 576, 64);
    gemm(stream, fp2cat, fp2w0, fp2a, nullptr, F(19), F(22), F(21),
         (int)nBN, 256, 576, 0, 0, 0, 1, 1);
    gemm(stream, fp2a, fp2w1, fp2b, nullptr, F(20), F(24), F(23),
         (int)nBN, 128, 256, 0, 0, 0, 1, 1);

    // ---- head ----
    gemm(stream, fp2b, w6f, h6, nullptr, F(0), F(8), F(7), (int)nBN, 64, 128, 0,0,0, 1, 1);
    k_add2<<<gsz(nBN*64), 256, 0, stream>>>(h6, f0, h6f, nullptr, 64, 0, 0, nBN*64);
    gemm(stream, h6f, w7f, h7, nullptr, F(1), F(10), F(9), (int)nBN, 32, 64, 0,0,0, 1, 1);
    gemm(stream, h7, wlf, nullptr, (float*)d_out, F(2), nullptr, nullptr,
         (int)nBN, 8, 32, 0, 0, 0, 0, 1);
}
